// Net_86801289052267
// MI455X (gfx1250) — matrix operand order repaired and checked by reading, not compiled, not run on hardware
//
#include <hip/hip_runtime.h>

// ---------------------------------------------------------------------------
// MI455X (gfx1250) implementation.
// Compute-bound bf16 WMMA GEMM chain (~350 GFLOP) vs ~0.4 GB HBM traffic
// (~17us at 23.3 TB/s) -> tensor-core bound. v_wmma_f32_16x16x32_bf16 with
// double-buffered LDS tiles filled by GLOBAL_LOAD_ASYNC_TO_LDS_B128
// (ASYNCcnt-tracked CDNA5 async copies), 8 wave32s per 128x128 block tile.
// ---------------------------------------------------------------------------

typedef __bf16 bf16_t;
typedef __bf16 v16bf __attribute__((ext_vector_type(16)));
typedef float  v8f   __attribute__((ext_vector_type(8)));

#define NC_PER_TASK 5
#define N_EXEMPLAR  20
#define N_CLASSES   100

static __device__ __forceinline__ bf16_t f32_to_bf16(float f) {
  union { float f; unsigned u; } v; v.f = f;
  unsigned r = v.u + 0x7FFFu + ((v.u >> 16) & 1u);   // round-to-nearest-even
  union { unsigned short s; bf16_t b; } o;
  o.s = (unsigned short)(r >> 16);
  return o.b;
}

// Low 32 bits of a generic pointer to __shared__ = LDS byte address.
static __device__ __forceinline__ unsigned lds_addr(const void* p) {
  return (unsigned)(uintptr_t)p;
}

// CDNA5 async global->LDS copy (ASYNCcnt-tracked, no VGPR round trip).
static __device__ __forceinline__ void async_copy_b128(unsigned lds,
                                                       const void* gptr) {
  asm volatile("global_load_async_to_lds_b128 %0, %1, off"
               :
               : "v"(lds), "v"(gptr)
               : "memory");
}

#define WAIT_ASYNCCNT(n) \
  asm volatile("s_wait_asynccnt %0" ::"i"(n) : "memory")

// ---------------------------------------------------------------------------
// W [K,N] f32 (row-major) -> Wt [Npad,K] bf16 (row-major), rows n>=N zeroed.
// ---------------------------------------------------------------------------
__global__ __launch_bounds__(256)
void transpose_to_bf16(const float* __restrict__ W, bf16_t* __restrict__ Wt,
                       int K, int N, int Npad) {
  __shared__ float tile[32][33];
  const int kb = blockIdx.x * 32;
  const int nb = blockIdx.y * 32;
  const int tx = threadIdx.x & 31;
  const int ty = threadIdx.x >> 5;  // 0..7
  for (int r = ty; r < 32; r += 8) {
    int k = kb + r, n = nb + tx;
    tile[r][tx] = (k < K && n < N) ? W[(size_t)k * N + n] : 0.0f;
  }
  __syncthreads();
  for (int r = ty; r < 32; r += 8) {
    int n = nb + r, k = kb + tx;
    if (n < Npad && k < K) Wt[(size_t)n * K + k] = f32_to_bf16(tile[tx][r]);
  }
}

// ---------------------------------------------------------------------------
// C[M,N] = act(A[M,K] * Wt[N,K]^T + bias)
//   A: bf16 row-major (f32 row-major if A_F32 -> converted during LDS fill)
//   Wt: bf16 [Npad,K] row-major, Npad = gridDim.x*128, padded rows are zero
//   BOUNDS: guard A rows against M (exemplar path, M=100); otherwise M must
//           be a multiple of 128 and A staging is fully async.
//   block: 256 thr = 8 wave32, 128x128 tile; wave = 64x32 (4x2 WMMA subtiles)
// ---------------------------------------------------------------------------
template <bool A_F32, bool BOUNDS, bool RELU, bool OUT_BF16>
__global__ __launch_bounds__(256)
void gemm_wmma(const void* __restrict__ Ain, const bf16_t* __restrict__ Bt,
               const float* __restrict__ bias, void* __restrict__ Cout,
               int M, int N, int K, int ldc) {
  constexpr int BK = 32;
  constexpr int LS = BK + 8;  // padded LDS stride (breaks bank conflicts)
  constexpr bool A_ASYNC = !A_F32 && !BOUNDS;
  // async instructions issued per tile per wave (in-order completion)
  constexpr int APT = A_ASYNC ? 4 : 2;

  __shared__ alignas(16) bf16_t As[2][128 * LS];
  __shared__ alignas(16) bf16_t Bs[2][128 * LS];

  const int tid  = threadIdx.x;
  const int lane = tid & 31;
  const int wave = tid >> 5;   // 0..7
  const int wm   = wave >> 2;  // 0..1  -> 64-row strip
  const int wn   = wave & 3;   // 0..3  -> 32-col strip
  const int half = lane >> 4;  // 0..1
  const int lidx = lane & 15;

  const int blockM = blockIdx.y * 128;
  const int blockN = blockIdx.x * 128;

  // cooperative tile loader: each thread owns 16 contiguous K-elements
  const int lrow = tid >> 1;
  const int lcol = (tid & 1) * 16;

  const bf16_t* Abf = (const bf16_t*)Ain;
  const float*  Af  = (const float*)Ain;

  bf16_t* const dstA[2] = {As[0] + lrow * LS + lcol, As[1] + lrow * LS + lcol};
  bf16_t* const dstB[2] = {Bs[0] + lrow * LS + lcol, Bs[1] + lrow * LS + lcol};
  const unsigned ldsA[2] = {lds_addr(dstA[0]), lds_addr(dstA[1])};
  const unsigned ldsB[2] = {lds_addr(dstB[0]), lds_addr(dstB[1])};

  // ---- tile staging ----
  auto issue_tile = [&](int k0, int buf) {
    // B tile: rows always valid (Wt padded to grid width) -> async DMA
    {
      const bf16_t* g = Bt + (size_t)(blockN + lrow) * K + k0 + lcol;
      async_copy_b128(ldsB[buf], g);
      async_copy_b128(ldsB[buf] + 16, g + 8);
    }
    if (A_ASYNC) {
      const bf16_t* g = Abf + (size_t)(blockM + lrow) * K + k0 + lcol;
      async_copy_b128(ldsA[buf], g);
      async_copy_b128(ldsA[buf] + 16, g + 8);
    } else if (A_F32) {
      const int m = blockM + lrow;
      float tmp[16];
      if (!BOUNDS || m < M) {
        const float4* g = (const float4*)(Af + (size_t)m * K + k0 + lcol);
        if (k0 + BK < K) __builtin_prefetch(g + K / 4, 0, 3);
#pragma unroll
        for (int q = 0; q < 4; ++q) {
          float4 f = g[q];
          tmp[q * 4 + 0] = f.x; tmp[q * 4 + 1] = f.y;
          tmp[q * 4 + 2] = f.z; tmp[q * 4 + 3] = f.w;
        }
      } else {
#pragma unroll
        for (int q = 0; q < 16; ++q) tmp[q] = 0.0f;
      }
#pragma unroll
      for (int q = 0; q < 16; ++q) dstA[buf][q] = f32_to_bf16(tmp[q]);
    } else {  // bf16 + bounds-guarded (exemplar path)
      uint4 q0 = {0, 0, 0, 0}, q1 = {0, 0, 0, 0};
      const int m = blockM + lrow;
      if (m < M) {
        const uint4* g = (const uint4*)(Abf + (size_t)m * K + k0 + lcol);
        q0 = g[0];
        q1 = g[1];
      }
      ((uint4*)dstA[buf])[0] = q0;
      ((uint4*)dstA[buf])[1] = q1;
    }
  };

  v8f acc[4][2];
  const v8f vzero = {0.f, 0.f, 0.f, 0.f, 0.f, 0.f, 0.f, 0.f};
#pragma unroll
  for (int i = 0; i < 4; ++i)
#pragma unroll
    for (int j = 0; j < 2; ++j) acc[i][j] = vzero;

  issue_tile(0, 0);

  for (int k0 = 0, step = 0; k0 < K; k0 += BK, ++step) {
    const int buf = step & 1;
    const bool has_next = (k0 + BK) < K;
    if (has_next) issue_tile(k0 + BK, buf ^ 1);
    // wait own async ops of CURRENT tile (next tile's APT still in flight)
    if (has_next) WAIT_ASYNCCNT(APT);
    else          WAIT_ASYNCCNT(0);
    __syncthreads();  // all waves' tile fragments visible

    // ---- fragments per ISA layout + 8 WMMAs ----
    union Frag { v16bf v; uint4 q[2]; };
    Frag afrag[4], bfrag[2];
#pragma unroll
    for (int i = 0; i < 4; ++i) {
      // A 16x32 bf16: lane-half0 K 0-7 & 16-23, half1 K 8-15 & 24-31
      const bf16_t* base = As[buf] + (wm * 64 + i * 16 + lidx) * LS + half * 8;
      afrag[i].q[0] = *(const uint4*)(base);       // K = half*8 .. +7
      afrag[i].q[1] = *(const uint4*)(base + 16);  // K = 16+half*8 .. +7
    }
#pragma unroll
    for (int j = 0; j < 2; ++j) {
      // B 32x16: lane-half0 K 0-7 & 16-23, half1 K 8-15 & 24-31
      const bf16_t* base = Bs[buf] + (wn * 32 + j * 16 + lidx) * LS + half * 8;
      bfrag[j].q[0] = *(const uint4*)(base);
      bfrag[j].q[1] = *(const uint4*)(base + 16);
    }
#pragma unroll
    for (int i = 0; i < 4; ++i)
#pragma unroll
      for (int j = 0; j < 2; ++j)
        acc[i][j] = __builtin_amdgcn_wmma_f32_16x16x32_bf16(
            false, afrag[i].v, false, bfrag[j].v, (short)0, acc[i][j], false,
            false);
    __syncthreads();  // protect buf before it is re-staged next iteration
  }

  // ---- epilogue: C VGPR r -> row half*8+r, col = lane&15 ----
#pragma unroll
  for (int i = 0; i < 4; ++i) {
    const int mbase = blockM + wm * 64 + i * 16 + half * 8;
#pragma unroll
    for (int j = 0; j < 2; ++j) {
      const int col = blockN + wn * 32 + j * 16 + lidx;
      if (col >= N) continue;
      const float bv = bias[col];
#pragma unroll
      for (int r = 0; r < 8; ++r) {
        const int row = mbase + r;
        if (BOUNDS && row >= M) continue;
        float v = acc[i][j][r] + bv;
        if (RELU) v = v > 0.0f ? v : 0.0f;
        if (OUT_BF16)
          ((bf16_t*)Cout)[(size_t)row * ldc + col] = f32_to_bf16(v);
        else
          ((float*)Cout)[(size_t)row * ldc + col] = v;
      }
    }
  }
}

// ---------------------------------------------------------------------------
// means[c][j] = mean_e mpreds[(c*20+e)][j]   (stride-128 rows, padded zeros)
// ---------------------------------------------------------------------------
__global__ void reduce_means(const float* __restrict__ mpreds,
                             float* __restrict__ means) {
  const int c = blockIdx.x;   // 0..4
  const int j = threadIdx.x;  // 0..127
  float s = 0.0f;
  if (j < N_CLASSES)
    for (int e = 0; e < N_EXEMPLAR; ++e)
      s += mpreds[(size_t)(c * N_EXEMPLAR + e) * 128 + j];
  means[c * 128 + j] = (j < N_CLASSES) ? s * (1.0f / N_EXEMPLAR) : 0.0f;
}

// ---------------------------------------------------------------------------
// per-sample: argmin_c ||means[c]-preds[s]||^2, one-hot at (argmin + t*5)
// ---------------------------------------------------------------------------
__global__ __launch_bounds__(256)
void argmin_onehot(const float* __restrict__ preds,
                   const float* __restrict__ means,
                   const int* __restrict__ tptr, float* __restrict__ out,
                   int ns) {
  __shared__ float sm[NC_PER_TASK * 128];
  for (int i = threadIdx.x; i < NC_PER_TASK * 128; i += blockDim.x)
    sm[i] = means[i];
  __syncthreads();
  const int s = blockIdx.x * blockDim.x + threadIdx.x;
  if (s >= ns) return;
  const float* p = preds + (size_t)s * 128;
  float d[NC_PER_TASK] = {0.f, 0.f, 0.f, 0.f, 0.f};
  for (int j = 0; j < N_CLASSES; ++j) {
    const float pv = p[j];
#pragma unroll
    for (int c = 0; c < NC_PER_TASK; ++c) {
      const float df = sm[c * 128 + j] - pv;
      d[c] += df * df;
    }
  }
  int best = 0;
  float bv = d[0];
#pragma unroll
  for (int c = 1; c < NC_PER_TASK; ++c)
    if (d[c] < bv) { bv = d[c]; best = c; }  // strict < keeps first (argmin)
  const int cls = best + tptr[0] * NC_PER_TASK;
  float* o = out + (size_t)s * N_CLASSES;
  for (int j = 0; j < N_CLASSES; ++j) o[j] = (j == cls) ? 1.0f : 0.0f;
}

// ---------------------------------------------------------------------------
extern "C" void kernel_launch(void* const* d_in, const int* in_sizes, int n_in,
                              void* d_out, int out_size, void* d_ws,
                              size_t ws_size, hipStream_t stream) {
  const float* x    = (const float*)d_in[0];  // [16384,3072]
  const float* memx = (const float*)d_in[1];  // [5,20,3072]
  const float* W1   = (const float*)d_in[2];  // [3072,2048]
  const float* b1   = (const float*)d_in[3];
  const float* W2   = (const float*)d_in[4];  // [2048,2048]
  const float* b2   = (const float*)d_in[5];
  const float* W3   = (const float*)d_in[6];  // [2048,100]
  const float* b3   = (const float*)d_in[7];
  const int*   t    = (const int*)d_in[8];

  const int NS = 16384, DIN = 3072, DH = 2048;
  const int MMEM = NC_PER_TASK * N_EXEMPLAR;  // 100
  const int NPAD3 = 128;

  char* w = (char*)d_ws;
  auto salloc = [&](size_t bytes) {
    char* p = w;
    w += (bytes + 255) & ~(size_t)255;
    return p;
  };
  bf16_t* W1t   = (bf16_t*)salloc((size_t)DH * DIN * 2);
  bf16_t* W2t   = (bf16_t*)salloc((size_t)DH * DH * 2);
  bf16_t* W3t   = (bf16_t*)salloc((size_t)NPAD3 * DH * 2);
  bf16_t* h1    = (bf16_t*)salloc((size_t)NS * DH * 2);
  bf16_t* h2    = (bf16_t*)salloc((size_t)NS * DH * 2);
  float*  preds = (float*)salloc((size_t)NS * NPAD3 * 4);
  bf16_t* mh1   = (bf16_t*)salloc((size_t)MMEM * DH * 2);
  bf16_t* mh2   = (bf16_t*)salloc((size_t)MMEM * DH * 2);
  float*  mpred = (float*)salloc((size_t)MMEM * NPAD3 * 4);
  float*  means = (float*)salloc((size_t)NC_PER_TASK * 128 * 4);
  (void)ws_size; (void)n_in; (void)in_sizes; (void)out_size;

  // weight transposes (bf16, [N,K], zero-padded rows for layer 3)
  transpose_to_bf16<<<dim3((DIN + 31) / 32, (DH + 31) / 32), 256, 0, stream>>>(
      W1, W1t, DIN, DH, DH);
  transpose_to_bf16<<<dim3((DH + 31) / 32, (DH + 31) / 32), 256, 0, stream>>>(
      W2, W2t, DH, DH, DH);
  transpose_to_bf16<<<dim3((DH + 31) / 32, (NPAD3 + 31) / 32), 256, 0,
                      stream>>>(W3, W3t, DH, N_CLASSES, NPAD3);

  // main path: x -> h1 -> h2 -> preds (M multiple of 128 -> unguarded/async)
  gemm_wmma<true, false, true, true>
      <<<dim3(DH / 128, NS / 128), 256, 0, stream>>>(x, W1t, b1, h1, NS, DH,
                                                     DIN, DH);
  gemm_wmma<false, false, true, true>
      <<<dim3(DH / 128, NS / 128), 256, 0, stream>>>(h1, W2t, b2, h2, NS, DH,
                                                     DH, DH);
  gemm_wmma<false, false, false, false>
      <<<dim3(NPAD3 / 128, NS / 128), 256, 0, stream>>>(h2, W3t, b3, preds, NS,
                                                        N_CLASSES, DH, NPAD3);

  // exemplar path: mem_x (100 rows) -> mpred (bounds-guarded A staging)
  gemm_wmma<true, true, true, true>
      <<<dim3(DH / 128, 1), 256, 0, stream>>>(memx, W1t, b1, mh1, MMEM, DH,
                                              DIN, DH);
  gemm_wmma<false, true, true, true>
      <<<dim3(DH / 128, 1), 256, 0, stream>>>(mh1, W2t, b2, mh2, MMEM, DH, DH,
                                              DH);
  gemm_wmma<false, true, false, false>
      <<<dim3(1, 1), 256, 0, stream>>>(mh2, W3t, b3, mpred, MMEM, N_CLASSES,
                                       DH, NPAD3);

  reduce_means<<<NC_PER_TASK, 128, 0, stream>>>(mpred, means);
  argmin_onehot<<<(NS + 255) / 256, 256, 0, stream>>>(preds, means, t,
                                                      (float*)d_out, NS);
}
